// HGT_10557029613960
// MI455X (gfx1250) — matrix operand order repaired and checked by reading, not compiled, not run on hardware
//
#include <hip/hip_runtime.h>
#include <hip/hip_bf16.h>
#include <math.h>

// ---------------------------------------------------------------------------
// HGT (heterogeneous graph transformer) for MI455X / gfx1250.
// Dense GEMMs -> v_wmma_f32_16x16x32_bf16 (bf16 operands, f32 accum),
// double-buffered LDS staging, batched B-fragment loads so WMMAs issue
// back-to-back. Weights pre-converted to transposed bf16 (Wt[n][k]).
// Edge softmax/scatter -> global atomics. BN -> 2-stage column reduction.
// ---------------------------------------------------------------------------

typedef __bf16 bf16;
typedef __attribute__((ext_vector_type(16))) __bf16 v16bf;
typedef __attribute__((ext_vector_type(8)))  float  v8f;

#define NBUS  (19 * 8192)
#define NGEN  (4 * 8192)
#define NLOAD (8 * 8192)
#define HID   128
#define NHEAD 8
#define HD    16
#define MLPM  (4 * 8192)   // 32768 rows into the MLP head

enum { EPI_NONE = 0, EPI_BIAS = 1, EPI_BIAS_RELU = 2, EPI_SKIP = 3 };

// ---------------- monotonic float<->uint encoding for atomicMax -------------
__device__ __forceinline__ unsigned fenc(float f) {
  unsigned u = __float_as_uint(f);
  return (u & 0x80000000u) ? ~u : (u | 0x80000000u);
}
__device__ __forceinline__ float fdec(unsigned u) {
  return (u & 0x80000000u) ? __uint_as_float(u & 0x7FFFFFFFu)
                           : __uint_as_float(~u);
}

// ---------------------------------------------------------------------------
// WMMA GEMM:  C[M,128] = epilogue( A[M,Ka](f32) @ W[Ka,128] )
// W supplied transposed bf16: Wt[n*Kp + k], n in [0,128), k in [0,Kp).
// 256 threads = 8 waves; block tile 128x128; K-step 32; double-buffered LDS.
// Wave w computes rows [w*16, w*16+16) x all 128 cols = 8 wmma accumulators.
// FULLK: Ka == Kp and M % 128 == 0 -> unchecked vectorized staging.
// ---------------------------------------------------------------------------
template <int EPI, bool FULLK>
__global__ __launch_bounds__(256) void gemm_bf16(
    const float* __restrict__ A, int M, int Ka, int Kp,
    const bf16* __restrict__ Wt, const float* __restrict__ bias,
    float* __restrict__ C, const float* __restrict__ skipPtr,
    const float* __restrict__ Xold) {
  __shared__ bf16 As[2][128][40];  // [buf][row][k], stride 80B (16B runs)
  __shared__ bf16 Bs[2][128][40];  // [buf][col][k]

  const int t = threadIdx.x;
  const int lane = t & 31, wave = t >> 5;
  const int half = lane >> 4, l15 = lane & 15;
  const int rowTile = blockIdx.x * 128;

  union AF { v16bf v; bf16 e[16]; };
  union CF { v8f v; float e[8]; };
  union PK { bf16 e[16]; uint4 q[2]; };
  CF acc[8];
#pragma unroll
  for (int j = 0; j < 8; j++)
#pragma unroll
    for (int r = 0; r < 8; r++) acc[j].e[r] = 0.f;

  const int rHalf = t >> 1;           // 0..127
  const int kOff = (t & 1) * 16;      // 0 or 16
  const int m = wave * 16 + l15;

  if (FULLK) {
    const float* aBase = A + (size_t)(rowTile + rHalf) * Ka + kOff;
    const bf16* bBase = Wt + (size_t)rHalf * Kp + kOff;
    // ---- prologue: stage K-step 0 into buffer 0 ----
    {
      PK pk;
#pragma unroll
      for (int i = 0; i < 4; i++) {
        float4 f = ((const float4*)aBase)[i];
        pk.e[4 * i + 0] = (bf16)f.x; pk.e[4 * i + 1] = (bf16)f.y;
        pk.e[4 * i + 2] = (bf16)f.z; pk.e[4 * i + 3] = (bf16)f.w;
      }
      *(uint4*)&As[0][rHalf][kOff] = pk.q[0];
      *(uint4*)&As[0][rHalf][kOff + 8] = pk.q[1];
      uint4 q0 = ((const uint4*)bBase)[0];
      uint4 q1 = ((const uint4*)bBase)[1];
      *(uint4*)&Bs[0][rHalf][kOff] = q0;
      *(uint4*)&Bs[0][rHalf][kOff + 8] = q1;
    }
    __syncthreads();
    for (int kt = 0; kt < Kp; kt += 32) {
      const int cur = (kt >> 5) & 1;
      const bool more = (kt + 32) < Kp;
      // issue next tile's global loads early (overlap with compute)
      float4 fA[4]; uint4 qB0, qB1;
      if (more) {
        const float* ap = aBase + kt + 32;
#pragma unroll
        for (int i = 0; i < 4; i++) fA[i] = ((const float4*)ap)[i];
        const bf16* bp = bBase + kt + 32;
        qB0 = ((const uint4*)bp)[0];
        qB1 = ((const uint4*)bp)[1];
      }
      // ---- compute from buffer `cur`: batch all LDS loads, then 8 WMMAs ----
      AF a;
#pragma unroll
      for (int i = 0; i < 8; i++) {
        a.e[i]     = As[cur][m][half * 8 + i];
        a.e[8 + i] = As[cur][m][16 + half * 8 + i];
      }
      AF bfr[8];
#pragma unroll
      for (int j = 0; j < 8; j++) {
        int col = j * 16 + l15;
#pragma unroll
        for (int i = 0; i < 16; i++) bfr[j].e[i] = Bs[cur][col][(i < 8 ? half * 8 + i : 8 + half * 8 + i)];
      }
#pragma unroll
      for (int j = 0; j < 8; j++)
        acc[j].v = __builtin_amdgcn_wmma_f32_16x16x32_bf16(
            false, a.v, false, bfr[j].v, (short)0, acc[j].v, false, false);
      // ---- stage next tile into the other buffer ----
      if (more) {
        PK pk;
#pragma unroll
        for (int i = 0; i < 4; i++) {
          pk.e[4 * i + 0] = (bf16)fA[i].x; pk.e[4 * i + 1] = (bf16)fA[i].y;
          pk.e[4 * i + 2] = (bf16)fA[i].z; pk.e[4 * i + 3] = (bf16)fA[i].w;
        }
        *(uint4*)&As[cur ^ 1][rHalf][kOff] = pk.q[0];
        *(uint4*)&As[cur ^ 1][rHalf][kOff + 8] = pk.q[1];
        *(uint4*)&Bs[cur ^ 1][rHalf][kOff] = qB0;
        *(uint4*)&Bs[cur ^ 1][rHalf][kOff + 8] = qB1;
      }
      __syncthreads();
    }
  } else {
    // generic path (input layer: Ka in {16,10,4}, Kp = 32, single K-step)
    for (int kt = 0; kt < Kp; kt += 32) {
      __syncthreads();
#pragma unroll
      for (int i = 0; i < 16; i++) {
        int idx = t * 16 + i;
        int r = idx >> 5, kk = idx & 31;
        int gr = rowTile + r, gk = kt + kk;
        float v = (gr < M && gk < Ka) ? A[(size_t)gr * Ka + gk] : 0.f;
        As[0][r][kk] = (bf16)v;
      }
#pragma unroll
      for (int i = 0; i < 16; i++) {
        int idx = t * 16 + i;
        int n = idx >> 5, kk = idx & 31;
        Bs[0][n][kk] = Wt[(size_t)n * Kp + kt + kk];
      }
      __syncthreads();
      AF a;
#pragma unroll
      for (int i = 0; i < 8; i++) {
        a.e[i]     = As[0][m][half * 8 + i];
        a.e[8 + i] = As[0][m][16 + half * 8 + i];
      }
      AF bfr[8];
#pragma unroll
      for (int j = 0; j < 8; j++) {
        int col = j * 16 + l15;
#pragma unroll
        for (int i = 0; i < 16; i++) bfr[j].e[i] = Bs[0][col][(i < 8 ? half * 8 + i : 8 + half * 8 + i)];
      }
#pragma unroll
      for (int j = 0; j < 8; j++)
        acc[j].v = __builtin_amdgcn_wmma_f32_16x16x32_bf16(
            false, a.v, false, bfr[j].v, (short)0, acc[j].v, false, false);
    }
  }

  float alpha = 0.f;
  if (EPI == EPI_SKIP) alpha = 1.f / (1.f + __expf(-*skipPtr));
#pragma unroll
  for (int j = 0; j < 8; j++) {
    int col = j * 16 + l15;
    float bv = (EPI == EPI_NONE) ? 0.f : bias[col];
#pragma unroll
    for (int r = 0; r < 8; r++) {
      int row = rowTile + wave * 16 + half * 8 + r;  // C/D: VGPR r -> M=r (+8 hi half)
      if (FULLK || row < M) {
        float v = acc[j].e[r] + bv;
        if (EPI == EPI_BIAS_RELU) v = fmaxf(v, 0.f);
        if (EPI == EPI_SKIP)
          v = alpha * v + (1.f - alpha) * Xold[(size_t)row * 128 + col];
        C[(size_t)row * 128 + col] = v;
      }
    }
  }
}

// ---------------------------- small utility kernels ------------------------
__global__ void fill_f32(float* p, size_t n, float v) {
  size_t i = (size_t)blockIdx.x * blockDim.x + threadIdx.x;
  if (i < n) p[i] = v;
}
__global__ void fill_u32(unsigned* p, size_t n, unsigned v) {
  size_t i = (size_t)blockIdx.x * blockDim.x + threadIdx.x;
  if (i < n) p[i] = v;
}
// W (K,128) f32 row-major -> Wt (128,Kp) bf16, zero-padded in k.
__global__ void cvt_padT(const float* __restrict__ W, int K, int Kp,
                         bf16* __restrict__ out) {
  int i = blockIdx.x * blockDim.x + threadIdx.x;
  if (i >= Kp * 128) return;
  int n = i / Kp, k = i % Kp;
  out[i] = (k < K) ? (bf16)W[(size_t)k * 128 + n] : (bf16)0.f;
}
// a_rel/m_rel (H,D,D) f32 -> block-diagonal transposed bf16 Wt[128][128].
__global__ void blockdiagT(const float* __restrict__ rel, bf16* __restrict__ out) {
  int i = blockIdx.x * blockDim.x + threadIdx.x;
  if (i >= 128 * 128) return;
  int n = i >> 7, k = i & 127;  // out[n][k] = W[k][n] = rel[h][k&15][n&15]
  out[i] = ((n >> 4) == (k >> 4))
               ? (bf16)rel[(n >> 4) * 256 + (k & 15) * 16 + (n & 15)]
               : (bf16)0.f;
}
__global__ void gelu_inplace(float* p, size_t n) {
  size_t i = (size_t)blockIdx.x * blockDim.x + threadIdx.x;
  if (i >= n) return;
  float x = p[i];
  p[i] = 0.5f * x * (1.f + erff(x * 0.70710678118654752f));
}
__global__ void min_fold(float* acc, const float* x, size_t n) {
  size_t i = (size_t)blockIdx.x * blockDim.x + threadIdx.x;
  if (i < n) acc[i] = fminf(acc[i], x[i]);
}

// ----------------------------- edge-stage kernels --------------------------
__global__ void edge_score(const int* __restrict__ src, const int* __restrict__ dst,
                           int E, const float* __restrict__ Q,
                           const float* __restrict__ KK,
                           const float* __restrict__ prel,
                           float* __restrict__ score, unsigned* __restrict__ mx) {
  int i = blockIdx.x * blockDim.x + threadIdx.x;
  if (i >= E * NHEAD) return;
  int e = i >> 3, h = i & 7;
  int s = src[e], d = dst[e];
  const float4* qp = (const float4*)(Q + (size_t)d * HID + h * HD);
  const float4* kp = (const float4*)(KK + (size_t)s * HID + h * HD);
  float acc = 0.f;
#pragma unroll
  for (int k = 0; k < 4; k++) {
    float4 q4 = qp[k], k4 = kp[k];
    acc += q4.x * k4.x + q4.y * k4.y + q4.z * k4.z + q4.w * k4.w;
  }
  acc *= prel[h] * 0.25f;  // 1/sqrt(D), D=16
  score[i] = acc;
  atomicMax(&mx[(size_t)d * NHEAD + h], fenc(acc));
}
__global__ void edge_expsum(const int* __restrict__ dst, int E,
                            float* __restrict__ score,
                            const unsigned* __restrict__ mx,
                            float* __restrict__ sum) {
  int i = blockIdx.x * blockDim.x + threadIdx.x;
  if (i >= E * NHEAD) return;
  int e = i >> 3, h = i & 7;
  int d = dst[e];
  float w = __expf(score[i] - fdec(mx[(size_t)d * NHEAD + h]));
  score[i] = w;
  atomicAdd(&sum[(size_t)d * NHEAD + h], w);
}
__global__ void edge_scatter(const int* __restrict__ src, const int* __restrict__ dst,
                             int E, const float* __restrict__ VV,
                             const float* __restrict__ score,
                             const float* __restrict__ sum, float* __restrict__ out) {
  int i = blockIdx.x * blockDim.x + threadIdx.x;
  if (i >= E * NHEAD) return;
  int e = i >> 3, h = i & 7;
  int s = src[e], d = dst[e];
  float a = score[i] / sum[(size_t)d * NHEAD + h];
  const float* vp = VV + (size_t)s * HID + h * HD;
  float* op = out + (size_t)d * HID + h * HD;
#pragma unroll
  for (int k = 0; k < HD; k++) atomicAdd(&op[k], vp[k] * a);
}

// ------------------------------- MLP head ----------------------------------
__global__ void gather_nodes(const float* __restrict__ Xb,
                             const int* __restrict__ nidx, float* __restrict__ H) {
  size_t i = (size_t)blockIdx.x * blockDim.x + threadIdx.x;
  if (i >= (size_t)MLPM * 128) return;
  int row = (int)(i >> 7), c = (int)(i & 127);
  int s = row >> 2, j = row & 3;
  int srcRow = 19 * s + nidx[j];
  H[i] = Xb[(size_t)srcRow * 128 + c];
}
__global__ void bn_stats(const float* __restrict__ X, int M, float* __restrict__ sums) {
  int c = threadIdx.x;  // 128 threads, one column each
  int per = (M + gridDim.x - 1) / gridDim.x;
  int r0 = blockIdx.x * per;
  int r1 = r0 + per; if (r1 > M) r1 = M;
  float s = 0.f, q = 0.f;
  for (int r = r0; r < r1; r++) {
    float v = X[(size_t)r * 128 + c];
    s += v; q += v * v;
  }
  atomicAdd(&sums[c], s);
  atomicAdd(&sums[128 + c], q);
}
__global__ void bn_apply_relu(float* __restrict__ X, int M,
                              const float* __restrict__ sums,
                              const float* __restrict__ gamma,
                              const float* __restrict__ beta) {
  size_t i = (size_t)blockIdx.x * blockDim.x + threadIdx.x;
  if (i >= (size_t)M * 128) return;
  int c = (int)(i & 127);
  float mu = sums[c] / (float)M;
  float var = sums[128 + c] / (float)M - mu * mu;
  float v = (X[i] - mu) * rsqrtf(var + 1e-5f) * gamma[c] + beta[c];
  X[i] = fmaxf(v, 0.f);
}
__global__ void final_lin(const float* __restrict__ H, const float* __restrict__ W,
                          const float* __restrict__ B, float* __restrict__ out) {
  int i = blockIdx.x * blockDim.x + threadIdx.x;
  if (i >= MLPM * 2) return;
  int r = i >> 1, o = i & 1;
  float acc = B[o];
#pragma unroll 4
  for (int k = 0; k < 128; k++) acc += H[(size_t)r * 128 + k] * W[k * 2 + o];
  out[i] = acc;
}

// ------------------------------- host side ---------------------------------
static void launch_gemm(int epi, bool fullk, const float* A, int M, int Ka, int Kp,
                        const bf16* Wt, const float* bias, float* C,
                        const float* skipPtr, const float* Xold, hipStream_t s) {
  dim3 grid((M + 127) / 128), blk(256);
  if (fullk) {
    switch (epi) {
      case EPI_NONE:      gemm_bf16<EPI_NONE, true><<<grid, blk, 0, s>>>(A, M, Ka, Kp, Wt, bias, C, skipPtr, Xold); break;
      case EPI_BIAS:      gemm_bf16<EPI_BIAS, true><<<grid, blk, 0, s>>>(A, M, Ka, Kp, Wt, bias, C, skipPtr, Xold); break;
      case EPI_BIAS_RELU: gemm_bf16<EPI_BIAS_RELU, true><<<grid, blk, 0, s>>>(A, M, Ka, Kp, Wt, bias, C, skipPtr, Xold); break;
      default:            gemm_bf16<EPI_SKIP, true><<<grid, blk, 0, s>>>(A, M, Ka, Kp, Wt, bias, C, skipPtr, Xold); break;
    }
  } else {
    gemm_bf16<EPI_BIAS_RELU, false><<<grid, blk, 0, s>>>(A, M, Ka, Kp, Wt, bias, C, skipPtr, Xold);
  }
}
static inline dim3 g1(size_t n, int t = 256) { return dim3((unsigned)((n + t - 1) / t)); }

extern "C" void kernel_launch(void* const* d_in, const int* in_sizes, int n_in,
                              void* d_out, int out_size, void* d_ws, size_t ws_size,
                              hipStream_t stream) {
  (void)in_sizes; (void)n_in; (void)out_size; (void)ws_size;
  const int Nn[3] = {NBUS, NGEN, NLOAD};
  const int FEAT[3] = {16, 10, 4};

  // ---- input layout --------------------------------------------------------
  // 0..2: x_bus/x_gen/x_load, 3..12: (src,dst) per et in bb,gb,bg,lb,bl order,
  // 13: node_idx, 14: num_graphs, then params flattened jax-pytree style
  // (sorted keys): bn(beta/gamma x4), convs[2](a_lin,a_rel,k,m_rel,p_rel,q,
  // skip,v with nt keys sorted bus/gen/load, et keys sorted bb,bg,bl,gb,lb,
  // {b,w} sorted), lin, mlp.
  const int P = 15;
  const int CONV0 = P + 8, CONVSTRIDE = 42;
  const int O_ALIN = 0, O_AREL = 6, O_K = 11, O_MREL = 17, O_PREL = 22,
            O_Q = 27, O_SKIP = 33, O_V = 36;
  const int LIN0 = P + 92;   // per nt: b, w
  const int MLP0 = P + 98;   // per i:  b, w

  struct ETm { int srcIn, dstIn, E, st, dt, sortIdx; };
  const ETm ets[5] = {
      {3, 4, 409600, 0, 0, 0},   // bb
      {5, 6, 32768, 1, 0, 3},    // gb
      {7, 8, 32768, 0, 1, 1},    // bg
      {9, 10, 65536, 2, 0, 4},   // lb
      {11, 12, 65536, 0, 2, 2},  // bl
  };

  // ---- workspace carve -----------------------------------------------------
  char* wp = (char*)d_ws;
  auto bumpf = [&](size_t e) { float* r = (float*)wp; wp += ((e * 4 + 255) / 256) * 256; return r; };
  auto bumpb = [&](size_t e) { bf16* r = (bf16*)wp; wp += ((e * 2 + 255) / 256) * 256; return r; };

  float *xc[3], *xn[3], *Qb[3], *Kb[3], *Vb[3], *outMin[3];
  for (int nt = 0; nt < 3; nt++) xc[nt] = bumpf((size_t)Nn[nt] * 128);
  for (int nt = 0; nt < 3; nt++) xn[nt] = bumpf((size_t)Nn[nt] * 128);
  for (int nt = 0; nt < 3; nt++) Qb[nt] = bumpf((size_t)Nn[nt] * 128);
  for (int nt = 0; nt < 3; nt++) Kb[nt] = bumpf((size_t)Nn[nt] * 128);
  for (int nt = 0; nt < 3; nt++) Vb[nt] = bumpf((size_t)Nn[nt] * 128);
  for (int nt = 0; nt < 3; nt++) outMin[nt] = bumpf((size_t)Nn[nt] * 128);
  float* kkBuf = bumpf((size_t)NBUS * 128);
  float* vvBuf = bumpf((size_t)NBUS * 128);
  float* outEt = bumpf((size_t)NBUS * 128);
  float* scoreBuf = bumpf((size_t)409600 * NHEAD);
  float* maxBuf = bumpf((size_t)NBUS * NHEAD);
  float* sumBuf = bumpf((size_t)NBUS * NHEAD);
  float* hA = bumpf((size_t)MLPM * 128);
  float* hB = bumpf((size_t)MLPM * 128);
  float* bnSums = bumpf(256);
  bf16* wLin[3];
  for (int nt = 0; nt < 3; nt++) wLin[nt] = bumpb(32 * 128);
  bf16 *wK[2][3], *wQ[2][3], *wV[2][3], *wA[2][3], *wArel[2][5], *wMrel[2][5];
  for (int L = 0; L < 2; L++) {
    for (int nt = 0; nt < 3; nt++) {
      wK[L][nt] = bumpb(128 * 128); wQ[L][nt] = bumpb(128 * 128);
      wV[L][nt] = bumpb(128 * 128); wA[L][nt] = bumpb(128 * 128);
    }
    for (int et = 0; et < 5; et++) {
      wArel[L][et] = bumpb(128 * 128); wMrel[L][et] = bumpb(128 * 128);
    }
  }
  bf16* wMlp[4];
  for (int i = 0; i < 4; i++) wMlp[i] = bumpb(128 * 128);

  // ---- weight conversion (f32 -> bf16 transposed, K-padded / blockdiag) ----
  for (int nt = 0; nt < 3; nt++)
    cvt_padT<<<g1(32 * 128), 256, 0, stream>>>((const float*)d_in[LIN0 + 2 * nt + 1], FEAT[nt], 32, wLin[nt]);
  for (int L = 0; L < 2; L++) {
    int cb = CONV0 + L * CONVSTRIDE;
    for (int nt = 0; nt < 3; nt++) {
      cvt_padT<<<g1(128 * 128), 256, 0, stream>>>((const float*)d_in[cb + O_K + 2 * nt + 1], 128, 128, wK[L][nt]);
      cvt_padT<<<g1(128 * 128), 256, 0, stream>>>((const float*)d_in[cb + O_Q + 2 * nt + 1], 128, 128, wQ[L][nt]);
      cvt_padT<<<g1(128 * 128), 256, 0, stream>>>((const float*)d_in[cb + O_V + 2 * nt + 1], 128, 128, wV[L][nt]);
      cvt_padT<<<g1(128 * 128), 256, 0, stream>>>((const float*)d_in[cb + O_ALIN + 2 * nt + 1], 128, 128, wA[L][nt]);
    }
    for (int et = 0; et < 5; et++) {
      int si = ets[et].sortIdx;
      blockdiagT<<<g1(128 * 128), 256, 0, stream>>>((const float*)d_in[cb + O_AREL + si], wArel[L][et]);
      blockdiagT<<<g1(128 * 128), 256, 0, stream>>>((const float*)d_in[cb + O_MREL + si], wMrel[L][et]);
    }
  }
  for (int i = 0; i < 4; i++)
    cvt_padT<<<g1(128 * 128), 256, 0, stream>>>((const float*)d_in[MLP0 + 2 * i + 1], 128, 128, wMlp[i]);

  // ---- input projection: x = relu(x_raw @ W + b) ---------------------------
  for (int nt = 0; nt < 3; nt++)
    launch_gemm(EPI_BIAS_RELU, false, (const float*)d_in[nt], Nn[nt], FEAT[nt], 32,
                wLin[nt], (const float*)d_in[LIN0 + 2 * nt], xc[nt], nullptr, nullptr, stream);

  // ---- HGT conv layers -----------------------------------------------------
  for (int L = 0; L < 2; L++) {
    int cb = CONV0 + L * CONVSTRIDE;
    for (int nt = 0; nt < 3; nt++) {
      launch_gemm(EPI_BIAS, true, xc[nt], Nn[nt], 128, 128, wK[L][nt],
                  (const float*)d_in[cb + O_K + 2 * nt], Kb[nt], nullptr, nullptr, stream);
      launch_gemm(EPI_BIAS, true, xc[nt], Nn[nt], 128, 128, wQ[L][nt],
                  (const float*)d_in[cb + O_Q + 2 * nt], Qb[nt], nullptr, nullptr, stream);
      launch_gemm(EPI_BIAS, true, xc[nt], Nn[nt], 128, 128, wV[L][nt],
                  (const float*)d_in[cb + O_V + 2 * nt], Vb[nt], nullptr, nullptr, stream);
      fill_f32<<<g1((size_t)Nn[nt] * 128), 256, 0, stream>>>(outMin[nt], (size_t)Nn[nt] * 128, 3.0e38f);
    }
    for (int et = 0; et < 5; et++) {
      const ETm& m = ets[et];
      const int* src = (const int*)d_in[m.srcIn];
      const int* dst = (const int*)d_in[m.dstIn];
      // per-edge-type key/msg projections as block-diagonal GEMMs
      launch_gemm(EPI_NONE, true, Kb[m.st], Nn[m.st], 128, 128, wArel[L][et], nullptr, kkBuf, nullptr, nullptr, stream);
      launch_gemm(EPI_NONE, true, Vb[m.st], Nn[m.st], 128, 128, wMrel[L][et], nullptr, vvBuf, nullptr, nullptr, stream);
      size_t nd8 = (size_t)Nn[m.dt] * NHEAD, nd128 = (size_t)Nn[m.dt] * 128;
      fill_u32<<<g1(nd8), 256, 0, stream>>>((unsigned*)maxBuf, nd8, 0x007FFFFFu);  // encoded -inf
      fill_f32<<<g1(nd8), 256, 0, stream>>>(sumBuf, nd8, 0.f);
      fill_f32<<<g1(nd128), 256, 0, stream>>>(outEt, nd128, 0.f);
      size_t nEH = (size_t)m.E * NHEAD;
      edge_score<<<g1(nEH), 256, 0, stream>>>(src, dst, m.E, Qb[m.dt], kkBuf,
                                              (const float*)d_in[cb + O_PREL + m.sortIdx],
                                              scoreBuf, (unsigned*)maxBuf);
      edge_expsum<<<g1(nEH), 256, 0, stream>>>(dst, m.E, scoreBuf, (const unsigned*)maxBuf, sumBuf);
      edge_scatter<<<g1(nEH), 256, 0, stream>>>(src, dst, m.E, vvBuf, scoreBuf, sumBuf, outEt);
      min_fold<<<g1(nd128), 256, 0, stream>>>(outMin[m.dt], outEt, nd128);
    }
    for (int nt = 0; nt < 3; nt++) {
      size_t n = (size_t)Nn[nt] * 128;
      gelu_inplace<<<g1(n), 256, 0, stream>>>(outMin[nt], n);
      launch_gemm(EPI_SKIP, true, outMin[nt], Nn[nt], 128, 128, wA[L][nt],
                  (const float*)d_in[cb + O_ALIN + 2 * nt], xn[nt],
                  (const float*)d_in[cb + O_SKIP + nt], xc[nt], stream);
    }
    for (int nt = 0; nt < 3; nt++) { float* t = xc[nt]; xc[nt] = xn[nt]; xn[nt] = t; }
  }

  // ---- MLP head with training-mode BatchNorm -------------------------------
  gather_nodes<<<g1((size_t)MLPM * 128), 256, 0, stream>>>(xc[0], (const int*)d_in[13], hA);
  float* hcur = hA; float* hnxt = hB;
  for (int i = 0; i < 4; i++) {
    launch_gemm(EPI_BIAS, true, hcur, MLPM, 128, 128, wMlp[i],
                (const float*)d_in[MLP0 + 2 * i], hnxt, nullptr, nullptr, stream);
    fill_f32<<<g1(256), 256, 0, stream>>>(bnSums, 256, 0.f);
    bn_stats<<<dim3(128), dim3(128), 0, stream>>>(hnxt, MLPM, bnSums);
    bn_apply_relu<<<g1((size_t)MLPM * 128), 256, 0, stream>>>(
        hnxt, MLPM, bnSums, (const float*)d_in[P + 2 * i + 1], (const float*)d_in[P + 2 * i]);
    float* t = hcur; hcur = hnxt; hnxt = t;
  }
  final_lin<<<g1(MLPM * 2), 256, 0, stream>>>(hcur, (const float*)d_in[MLP0 + 9],
                                              (const float*)d_in[MLP0 + 8], (float*)d_out);
}
